// MultiHeadSelfAttention_46840913330256
// MI455X (gfx1250) — matrix operand order repaired and checked by reading, not compiled, not run on hardware
//
#include <hip/hip_runtime.h>

typedef __attribute__((ext_vector_type(16))) _Float16 v16h;
typedef __attribute__((ext_vector_type(8)))  _Float16 v8h;
typedef __attribute__((ext_vector_type(8)))  float    v8f;
typedef _Float16 h16;

#define D_MODEL 1024
#define NHEAD   16
#define DK      64
#define BATCH   4
#define SEQ     2048
#define MROWS   (BATCH * SEQ)       // 8192
#define LDSP    72                  // padded LDS row stride (halfs): 36-dword stride, bank-conflict free

// ---------------------------------------------------------------------------
// fp32 -> f16 conversion (32B load / 16B store per thread)
// ---------------------------------------------------------------------------
__global__ void cvt_f32_f16_v8(const float* __restrict__ src,
                               h16* __restrict__ dst, int n8) {
  int i = blockIdx.x * blockDim.x + threadIdx.x;
  if (i < n8) {
    const float4* s4 = (const float4*)src;
    float4 a = s4[2 * i];
    float4 b = s4[2 * i + 1];
    v8h o;
    o[0] = (h16)a.x; o[1] = (h16)a.y; o[2] = (h16)a.z; o[3] = (h16)a.w;
    o[4] = (h16)b.x; o[5] = (h16)b.y; o[6] = (h16)b.z; o[7] = (h16)b.w;
    ((v8h*)dst)[i] = o;
  }
}

// ---------------------------------------------------------------------------
// GEMM: Y[m,e] = (sum_d A[m,d] * W[e,d] + bias[e]) * scale
// A: [8192,1024] f16 row-major, W: [1024,1024] f16 row-major ([out,in])
// mode 0: write f16 head-split [B,H,S,DK]   mode 1: write f32 [m,e] (d_out)
// Block: 256 thr = 8 waves (4x2); block tile 256x128; wave tile 64x64.
// ---------------------------------------------------------------------------
__global__ __launch_bounds__(256)
void gemm_wmma(const h16* __restrict__ A, const h16* __restrict__ W,
               const float* __restrict__ bias, h16* __restrict__ outh,
               float* __restrict__ outf, float scale, int mode) {
  const int lane  = threadIdx.x & 31;
  const int wave  = threadIdx.x >> 5;
  const int wm    = wave >> 1;          // 0..3
  const int wn    = wave & 1;           // 0..1
  const int Mbase = blockIdx.y * 256 + wm * 64;
  const int Nbase = blockIdx.x * 128 + wn * 64;
  const int l16   = lane & 15;
  const int lhalf = lane >> 4;          // 0/1

  v8f zero = {};
  v8f acc[4][4];
#pragma unroll
  for (int i = 0; i < 4; i++)
#pragma unroll
    for (int j = 0; j < 4; j++) acc[i][j] = zero;

  for (int kk = 0; kk < D_MODEL; kk += 32) {
    v16h af[4], bf[4];
#pragma unroll
    for (int am = 0; am < 4; am++) {
      // A frag layout: lane holds row (l16), K segs at kk+lhalf*8 and kk+16+lhalf*8
      const h16* p = A + (size_t)(Mbase + am * 16 + l16) * D_MODEL + kk + lhalf * 8;
      v8h s0 = *(const v8h*)p;
      v8h s1 = *(const v8h*)(p + 16);
#pragma unroll
      for (int i = 0; i < 8; i++) { af[am][i] = s0[i]; af[am][i + 8] = s1[i]; }
    }
#pragma unroll
    for (int an = 0; an < 4; an++) {
      // B frag: K segs at kk+lhalf*8 and kk+16+lhalf*8 (same as A)
      const h16* p = W + (size_t)(Nbase + an * 16 + l16) * D_MODEL + kk + lhalf * 8;
      v8h s0 = *(const v8h*)p;
      v8h s1 = *(const v8h*)(p + 16);
#pragma unroll
      for (int i = 0; i < 8; i++) { bf[an][i] = s0[i]; bf[an][i + 8] = s1[i]; }
    }
#pragma unroll
    for (int am = 0; am < 4; am++)
#pragma unroll
      for (int an = 0; an < 4; an++)
        acc[am][an] = __builtin_amdgcn_wmma_f32_16x16x32_f16(
            false, af[am], false, bf[an], (short)0, acc[am][an], false, false);
  }

  // Epilogue. C/D layout: VGPR i holds row M = lhalf*8 + i, col N = l16.
#pragma unroll
  for (int an = 0; an < 4; an++) {
    int col = Nbase + an * 16 + l16;
    float bv = bias[col];
#pragma unroll
    for (int am = 0; am < 4; am++) {
#pragma unroll
      for (int i = 0; i < 8; i++) {
        int m = Mbase + am * 16 + lhalf * 8 + i;
        float v = (acc[am][an][i] + bv) * scale;
        if (mode == 0) {
          int b = m >> 11, s = m & (SEQ - 1);
          int h = col >> 6, dk = col & 63;
          outh[(((size_t)(b * NHEAD + h) * SEQ + s) << 6) + dk] = (h16)v;
        } else {
          outf[(size_t)m * D_MODEL + col] = v;
        }
      }
    }
  }
}

// ---------------------------------------------------------------------------
// Flash attention: one block per (qblock of 128 rows, b*h). 8 waves,
// each wave owns 16 query rows. K block (64 keys) is staged into LDS with
// gfx1250 async-to-LDS copies (ASYNCcnt); V^T is built via a register
// transpose so PV B-fragments are contiguous ds_load_b128s. Next block's
// K/V lines are prefetched (global_prefetch_b8) during staging.
// Q is pre-scaled by 1/sqrt(DK) in the projection epilogue.
// ---------------------------------------------------------------------------
__global__ __launch_bounds__(256)
void attn_wmma(const h16* __restrict__ Qp, const h16* __restrict__ Kp,
               const h16* __restrict__ Vp, const int* __restrict__ mask,
               h16* __restrict__ ctx) {
  __shared__ __attribute__((aligned(16))) h16 Kb[64 * LDSP];      // K block, row-major (padded)
  __shared__ __attribute__((aligned(16))) h16 Vt[64 * LDSP];      // V^T: [dk][key]
  __shared__ __attribute__((aligned(16))) h16 Pl[8 * 16 * LDSP];  // per-wave P tiles
  __shared__ float mb[64];                                        // mask bias

  const int lane  = threadIdx.x & 31;
  const int wave  = threadIdx.x >> 5;
  const int l16   = lane & 15;
  const int lhalf = lane >> 4;
  const int bh    = blockIdx.y;            // 0..63
  const int b     = bh >> 4;
  const int h     = bh & 15;
  const int q0    = blockIdx.x * 128 + wave * 16;

  const h16* Qbh = Qp + (size_t)bh * SEQ * DK;
  const h16* Kbh = Kp + (size_t)bh * SEQ * DK;
  const h16* Vbh = Vp + (size_t)bh * SEQ * DK;

  // Q A-fragments for k=[0,32) and [32,64): kept resident all kernel.
  v16h qf[2];
#pragma unroll
  for (int j = 0; j < 2; j++) {
    const h16* p = Qbh + (size_t)(q0 + l16) * DK + j * 32 + lhalf * 8;
    v8h s0 = *(const v8h*)p;
    v8h s1 = *(const v8h*)(p + 16);
#pragma unroll
    for (int i = 0; i < 8; i++) { qf[j][i] = s0[i]; qf[j][i + 8] = s1[i]; }
  }

  float mi[8], li[8];
  v8f zero = {};
  v8f oacc[4];
#pragma unroll
  for (int i = 0; i < 8; i++) { mi[i] = -3.0e38f; li[i] = 0.0f; }
#pragma unroll
  for (int n = 0; n < 4; n++) oacc[n] = zero;

  h16* Pw = Pl + wave * 16 * LDSP;

  // per-thread staging coordinates: key row r (0..63), dk col c (0/16/32/48)
  const int r = threadIdx.x >> 2;
  const int c = (threadIdx.x & 3) << 4;

  for (int key0 = 0; key0 < SEQ; key0 += 64) {
    __syncthreads();  // previous iteration done reading Kb/Vt
    {
      // --- K block: async copy global -> LDS (2 x 16B per thread) ---
      unsigned klds = (unsigned)(uintptr_t)(&Kb[r * LDSP + c]);
      unsigned koff = (unsigned)(((unsigned)(key0 + r) * DK + c) * 2u);
      asm volatile("global_load_async_to_lds_b128 %0, %1, %2"
                   :: "v"(klds), "v"(koff), "s"(Kbh) : "memory");
      asm volatile("global_load_async_to_lds_b128 %0, %1, %2 offset:16"
                   :: "v"(klds), "v"(koff), "s"(Kbh) : "memory");

      // --- V block: load + transpose into Vt (register transpose) ---
      const h16* vp = Vbh + (size_t)(key0 + r) * DK + c;
      v8h v0v = *(const v8h*)vp;
      v8h v1v = *(const v8h*)(vp + 8);
#pragma unroll
      for (int j = 0; j < 8; j++) {
        Vt[(c + j) * LDSP + r]     = v0v[j];
        Vt[(c + 8 + j) * LDSP + r] = v1v[j];
      }
      if (threadIdx.x < 64)
        mb[threadIdx.x] = (mask[b * SEQ + key0 + threadIdx.x] == 0) ? -1.0e9f : 0.0f;

      // --- prefetch next key block while this one is consumed ---
      if (key0 + 64 < SEQ) {
        __builtin_prefetch(Kbh + (size_t)(key0 + 64 + r) * DK + c, 0, 1);
        __builtin_prefetch(vp + (size_t)64 * DK, 0, 1);
      }
    }
    asm volatile("s_wait_asynccnt 0" ::: "memory");
    __syncthreads();

    // scores S = Q @ K^T  (16 rows x 64 keys, 4 N-tiles)
    v8f sc[4];
#pragma unroll
    for (int t = 0; t < 4; t++) {
      sc[t] = zero;
#pragma unroll
      for (int j = 0; j < 2; j++) {
        const h16* p = &Kb[(t * 16 + l16) * LDSP + j * 32 + lhalf * 8];
        v8h s0 = *(const v8h*)p;
        v8h s1 = *(const v8h*)(p + 16);
        v16h bfr;
#pragma unroll
        for (int i = 0; i < 8; i++) { bfr[i] = s0[i]; bfr[i + 8] = s1[i]; }
        sc[t] = __builtin_amdgcn_wmma_f32_16x16x32_f16(
            false, qf[j], false, bfr, (short)0, sc[t], false, false);
      }
    }

    // mask bias + row max (row M lives in one 16-lane half, VGPR i)
    float rm[8];
#pragma unroll
    for (int i = 0; i < 8; i++) rm[i] = -3.0e38f;
#pragma unroll
    for (int t = 0; t < 4; t++) {
      float mbv = mb[t * 16 + l16];
#pragma unroll
      for (int i = 0; i < 8; i++) {
        float v = sc[t][i] + mbv;
        sc[t][i] = v;
        rm[i] = fmaxf(rm[i], v);
      }
    }
#pragma unroll
    for (int i = 0; i < 8; i++) {
      float v = rm[i];
      v = fmaxf(v, __shfl_xor(v, 1));
      v = fmaxf(v, __shfl_xor(v, 2));
      v = fmaxf(v, __shfl_xor(v, 4));
      v = fmaxf(v, __shfl_xor(v, 8));
      rm[i] = v;
    }

    // online softmax update
    float rs[8];
#pragma unroll
    for (int i = 0; i < 8; i++) {
      float mnew = fmaxf(mi[i], rm[i]);
      float f = __expf(mi[i] - mnew);
      mi[i] = mnew;
      li[i] *= f;
#pragma unroll
      for (int n = 0; n < 4; n++) oacc[n][i] *= f;
      rs[i] = 0.0f;
    }
#pragma unroll
    for (int t = 0; t < 4; t++) {
#pragma unroll
      for (int i = 0; i < 8; i++) {
        float p = __expf(sc[t][i] - mi[i]);
        rs[i] += p;
        Pw[(lhalf * 8 + i) * LDSP + t * 16 + l16] = (h16)p;  // C-layout -> LDS
      }
    }
#pragma unroll
    for (int i = 0; i < 8; i++) {
      float v = rs[i];
      v += __shfl_xor(v, 1);
      v += __shfl_xor(v, 2);
      v += __shfl_xor(v, 4);
      v += __shfl_xor(v, 8);
      li[i] += v;
    }
    __syncthreads();  // P stores visible before A-layout reload

    // O += P @ V  (A-frags from Pw, B-frags from Vt)
    v16h pa[2];
#pragma unroll
    for (int j = 0; j < 2; j++) {
      const h16* p = &Pw[l16 * LDSP + j * 32 + lhalf * 8];
      v8h s0 = *(const v8h*)p;
      v8h s1 = *(const v8h*)(p + 16);
#pragma unroll
      for (int i = 0; i < 8; i++) { pa[j][i] = s0[i]; pa[j][i + 8] = s1[i]; }
    }
#pragma unroll
    for (int n = 0; n < 4; n++) {
#pragma unroll
      for (int j = 0; j < 2; j++) {
        const h16* p = &Vt[(n * 16 + l16) * LDSP + j * 32 + lhalf * 8];
        v8h s0 = *(const v8h*)p;
        v8h s1 = *(const v8h*)(p + 16);
        v16h bfr;
#pragma unroll
        for (int i = 0; i < 8; i++) { bfr[i] = s0[i]; bfr[i + 8] = s1[i]; }
        oacc[n] = __builtin_amdgcn_wmma_f32_16x16x32_f16(
            false, pa[j], false, bfr, (short)0, oacc[n], false, false);
      }
    }
  }

  // normalize + write context in [B,S,H,DK] (row-major [8192,1024]) f16
#pragma unroll
  for (int n = 0; n < 4; n++) {
#pragma unroll
    for (int i = 0; i < 8; i++) {
      int s = q0 + lhalf * 8 + i;
      float v = oacc[n][i] / li[i];
      ctx[(((size_t)(b * SEQ + s) * NHEAD + h) << 6) + n * 16 + l16] = (h16)v;
    }
  }
}

// ---------------------------------------------------------------------------
// Host-side launch
// ---------------------------------------------------------------------------
extern "C" void kernel_launch(void* const* d_in, const int* in_sizes, int n_in,
                              void* d_out, int out_size, void* d_ws, size_t ws_size,
                              hipStream_t stream) {
  const float* q    = (const float*)d_in[0];
  const float* k    = (const float*)d_in[1];
  const float* v    = (const float*)d_in[2];
  const int*   mask = (const int*)d_in[3];
  const float* Wq   = (const float*)d_in[4];
  const float* bq   = (const float*)d_in[5];
  const float* Wk   = (const float*)d_in[6];
  const float* bk   = (const float*)d_in[7];
  const float* Wv   = (const float*)d_in[8];
  const float* bv   = (const float*)d_in[9];
  const float* Wo   = (const float*)d_in[10];
  const float* bo   = (const float*)d_in[11];
  float* out = (float*)d_out;

  // workspace layout (f16 elements): 120 MB total
  h16* ws = (h16*)d_ws;
  size_t o = 0;
  const size_t NX = (size_t)MROWS * D_MODEL;    // 8M
  const size_t NW = (size_t)D_MODEL * D_MODEL;  // 1M
  h16* xq  = ws + o; o += NX;
  h16* xk  = ws + o; o += NX;
  h16* xv  = ws + o; o += NX;
  h16* wqh = ws + o; o += NW;
  h16* wkh = ws + o; o += NW;
  h16* wvh = ws + o; o += NW;
  h16* woh = ws + o; o += NW;
  h16* Qp  = ws + o; o += NX;   // [B,H,S,DK]
  h16* Kpp = ws + o; o += NX;
  h16* Vpp = ws + o; o += NX;
  h16* ctx = ws + o; o += NX;   // [B,S,H,DK] == [8192,1024]

  const int n8x = (int)(NX / 8);
  const int n8w = (int)(NW / 8);
  cvt_f32_f16_v8<<<(n8x + 255) / 256, 256, 0, stream>>>(q, xq, n8x);
  cvt_f32_f16_v8<<<(n8x + 255) / 256, 256, 0, stream>>>(k, xk, n8x);
  cvt_f32_f16_v8<<<(n8x + 255) / 256, 256, 0, stream>>>(v, xv, n8x);
  cvt_f32_f16_v8<<<(n8w + 255) / 256, 256, 0, stream>>>(Wq, wqh, n8w);
  cvt_f32_f16_v8<<<(n8w + 255) / 256, 256, 0, stream>>>(Wk, wkh, n8w);
  cvt_f32_f16_v8<<<(n8w + 255) / 256, 256, 0, stream>>>(Wv, wvh, n8w);
  cvt_f32_f16_v8<<<(n8w + 255) / 256, 256, 0, stream>>>(Wo, woh, n8w);

  dim3 gg(D_MODEL / 128, MROWS / 256);  // (8, 32)
  // Q is pre-scaled by 1/sqrt(DK) = 0.125 so attention skips the scale
  gemm_wmma<<<gg, 256, 0, stream>>>(xq, wqh, bq, Qp, nullptr, 0.125f, 0);
  gemm_wmma<<<gg, 256, 0, stream>>>(xk, wkh, bk, Kpp, nullptr, 1.0f, 0);
  gemm_wmma<<<gg, 256, 0, stream>>>(xv, wvh, bv, Vpp, nullptr, 1.0f, 0);

  dim3 ga(SEQ / 128, BATCH * NHEAD);    // (16, 64)
  attn_wmma<<<ga, 256, 0, stream>>>(Qp, Kpp, Vpp, mask, ctx);

  gemm_wmma<<<gg, 256, 0, stream>>>(ctx, woh, bo, nullptr, out, 1.0f, 1);
}
